// RefineNet_8203387535596
// MI455X (gfx1250) — matrix operand order repaired and checked by reading, not compiled, not run on hardware
//
#include <hip/hip_runtime.h>
#include <hip/hip_bf16.h>

typedef _Float16 half_t;
typedef __attribute__((ext_vector_type(16))) _Float16 v16h;
typedef __attribute__((ext_vector_type(8)))  float    v8f;
typedef __attribute__((ext_vector_type(8)))  unsigned int v8u;

#define DIMF 64
#define MDIMF 32
#define EINF 129
#define H1F 258
#define H1P 288
#define NTE1 17      // ew1 n-tiles (258 -> 272)
#define RTOT 4416    // 2*96*23 rows total
#define NLAYER 16

__device__ __forceinline__ float silu_f(float x) { return x / (1.0f + __expf(-x)); }

union frag_cvt { v16h h; v8u u; };

// ---- WMMA fragment loaders (CDNA5 ISA 7.12.2 layouts, wave32) ----

// A 16x32 f16 from LDS (row-major, leading dim ld), tile origin (m0,k0); packed 4B reads
__device__ __forceinline__ v16h load_a_lds(const half_t* base, int ld, int m0, int k0) {
  const int lane = threadIdx.x & 31;
  const int m = m0 + (lane & 15);
  const int kb = k0 + ((lane >> 4) << 3);
  frag_cvt a;
#pragma unroll
  for (int v = 0; v < 8; ++v) {
    const int k = (v < 4) ? (kb + 2 * v) : (kb + 16 + 2 * (v - 4));
    a.u[v] = *reinterpret_cast<const unsigned int*>(base + m * ld + k);
  }
  return a.h;
}

// A 16x32 f16 for edge_in = [f_i (64) | f_j (64)]: k<64 broadcast row i, k>=64 per-row j
__device__ __forceinline__ v16h load_a_edge(const half_t* fi, const half_t* fj, int m0, int k0) {
  const int lane = threadIdx.x & 31;
  const int m = m0 + (lane & 15);
  const int kb = k0 + ((lane >> 4) << 3);
  frag_cvt a;
#pragma unroll
  for (int v = 0; v < 8; ++v) {
    const int k = (v < 4) ? (kb + 2 * v) : (kb + 16 + 2 * (v - 4));
    const half_t* src = (k < DIMF) ? (fi + k) : (fj + m * DIMF + (k - DIMF));
    a.u[v] = *reinterpret_cast<const unsigned int*>(src);
  }
  return a.h;
}

// B 32x16 f16 from pre-swizzled global weights: [ks][nt][lane][16 halves] -> one 32B load
__device__ __forceinline__ v16h load_b_sw(const half_t* __restrict__ Bsw, int ks, int nt, int NT) {
  const int lane = threadIdx.x & 31;
  return *reinterpret_cast<const v16h*>(Bsw + (((size_t)(ks * NT + nt) * 32 + lane) << 4));
}

#define WMMA_F16(A, B, C) \
  __builtin_amdgcn_wmma_f32_16x16x32_f16(false, (A), false, (B), (short)0, (C), false, false)

// ---------------- weight pre-pack: row-major f32 [K][N] -> swizzled padded f16 ----------------
__global__ void pack_b_kernel(const float* __restrict__ src, half_t* __restrict__ dst,
                              int srcK, int srcN, int Kused, int ksteps, int NT) {
  const int layer = blockIdx.y;
  const int total = ksteps * NT * 512;
  const float* s = src + (size_t)layer * srcK * srcN;
  half_t* d = dst + (size_t)layer * total;
  for (int idx = blockIdx.x * blockDim.x + threadIdx.x; idx < total;
       idx += gridDim.x * blockDim.x) {
    const int h = idx & 15;
    const int lane = (idx >> 4) & 31;
    const int nt = (idx >> 9) % NT;
    const int ks = idx / (NT * 512);
    const int n = nt * 16 + (lane & 15);
    const int k = ks * 32 + ((lane >> 4) << 3) + (h < 8 ? h : h + 8);
    float v = 0.0f;
    if (k < Kused && n < srcN) v = s[(size_t)k * srcN + n];
    d[idx] = (half_t)v;
  }
}

__global__ void pad_vec_kernel(const float* __restrict__ src, float* __restrict__ dst,
                               int Nused, int Npad, int srcStride) {
  const int layer = blockIdx.y;
  for (int idx = blockIdx.x * blockDim.x + threadIdx.x; idx < Npad;
       idx += gridDim.x * blockDim.x)
    dst[(size_t)layer * Npad + idx] = (idx < Nused) ? src[(size_t)layer * srcStride + idx] : 0.0f;
}

// ---------------- embedding + initial coords ----------------
__global__ __launch_bounds__(256) void embed_kernel(
    const int* __restrict__ tokens, const float* __restrict__ cords,
    const float* __restrict__ tok_emb, const float* __restrict__ pos_emb,
    float* __restrict__ feats, float* __restrict__ coords) {
  const int r = blockIdx.x;             // residue 0..191  (b*96+l)
  const int t = tokens[r];
  const int mask = (t != 1) ? 1 : 0;    // PAD == 1
  for (int idx = threadIdx.x; idx < 23 * DIMF; idx += blockDim.x) {
    const int n = idx >> 6, k = idx & 63;
    const int pos = mask ? (n + 2) : 1; // cumsum closed form (+PAD)
    feats[((size_t)r * 23 + n) * DIMF + k] = tok_emb[t * DIMF + k] + pos_emb[pos * DIMF + k];
  }
  for (int idx = threadIdx.x; idx < 23 * 3; idx += blockDim.x)
    coords[(size_t)r * 69 + idx] = cords[(size_t)r * 69 + idx];
}

// ---------------- zero scratch ----------------
__global__ void zero_kernel(float* __restrict__ p, int n) {
  for (int i = blockIdx.x * blockDim.x + threadIdx.x; i < n; i += gridDim.x * blockDim.x)
    p[i] = 0.0f;
}

// ---------------- edge MLP + coord weights (one block per node i, chunk of j) ----------------
template <int NN, int NCHUNKS, int CHUNKM>
__global__ __launch_bounds__(256) void egnn_edge_kernel(
    const float* __restrict__ feats, const float* __restrict__ coords,
    const half_t* __restrict__ ew1p, const float* __restrict__ eb1p,
    const float* __restrict__ ew1dp,
    const half_t* __restrict__ ew2p, const float* __restrict__ eb2,
    const half_t* __restrict__ cw1p, const float* __restrict__ cb1,
    const float* __restrict__ cw2, const float* __restrict__ cb2,
    const float* __restrict__ ln_b,
    float* __restrict__ m_i, float* __restrict__ cdelta,
    int layer) {
  constexpr int Mp = CHUNKM * 16;
  constexpr bool FULL = (NCHUNKS * Mp == NN);   // chunking exactly tiles the graph
  __shared__ half_t fi_sh[DIMF];
  __shared__ half_t fj_sh[Mp * DIMF];
  __shared__ half_t h1_sh[Mp * H1P];
  __shared__ half_t mij_sh[Mp * MDIMF];
  __shared__ float  dist_sh[Mp];
  __shared__ float  rel_sh[Mp * 3];
  __shared__ float  w_sh[Mp];
  __shared__ float  msum[MDIMF];
  __shared__ float  cdsum[3];

  const int tid = threadIdx.x;
  const int lane = tid & 31;
  const int wv = tid >> 5;
  const int cjk = blockIdx.x % NCHUNKS;
  const int node_i = (blockIdx.x / NCHUNKS) % NN;
  const int g = blockIdx.x / (NCHUNKS * NN);
  const int j0 = cjk * Mp;

  const half_t* ew1pL = ew1p + (size_t)layer * (4 * NTE1 * 512);
  const float*  eb1pL = eb1p + (size_t)layer * (NTE1 * 16);
  const float*  ew1dpL = ew1dp + (size_t)layer * (NTE1 * 16);
  const half_t* ew2pL = ew2p + (size_t)layer * (9 * 2 * 512);
  const float*  eb2L = eb2 + (size_t)layer * MDIMF;
  const half_t* cw1pL = cw1p + (size_t)layer * (1 * 8 * 512);
  const float*  cb1L = cb1 + (size_t)layer * 128;
  const float*  cw2L = cw2 + (size_t)layer * 128;
  const float cb2v = cb2[layer];
  const float beta = ln_b[layer];

  // ---- init LDS ----
  for (int idx = tid; idx < Mp * H1P; idx += 256) h1_sh[idx] = (half_t)0.0f;
  if (tid < Mp) w_sh[tid] = cb2v;
  if (tid < MDIMF) msum[tid] = 0.0f;
  if (tid < 3) cdsum[tid] = 0.0f;

  const float* fI = feats + (size_t)(g * NN + node_i) * DIMF;
  for (int idx = tid; idx < DIMF; idx += 256) fi_sh[idx] = (half_t)fI[idx];
  for (int idx = tid; idx < Mp * DIMF; idx += 256) {
    const int m = idx >> 6, k = idx & 63, j = j0 + m;
    float v = 0.0f;
    if (FULL || j < NN) v = feats[(size_t)(g * NN + j) * DIMF + k];
    fj_sh[idx] = (half_t)v;
  }
  if (tid < Mp) {
    const int j = j0 + tid;
    float rx = 0.f, ry = 0.f, rz = 0.f, d = 0.f;
    if (FULL || j < NN) {
      const float* ci = coords + (size_t)(g * NN + node_i) * 3;
      const float* cj = coords + (size_t)(g * NN + j) * 3;
      rx = ci[0] - cj[0]; ry = ci[1] - cj[1]; rz = ci[2] - cj[2];
      d = rx * rx + ry * ry + rz * rz;
    }
    rel_sh[tid * 3 + 0] = rx; rel_sh[tid * 3 + 1] = ry; rel_sh[tid * 3 + 2] = rz;
    dist_sh[tid] = d;
  }
  __syncthreads();

  // ---- GEMM1: edge_in (Mp x 129) @ ew1 (129 x 258) -> h1, silu ----
  // K=128 via WMMA, row 128 (dist) as rank-1 VALU update; all pads are zero -> silu(0)=0.
  for (int nt = wv; nt < NTE1; nt += 8) {
    v8f acc[CHUNKM];
#pragma unroll
    for (int mt = 0; mt < CHUNKM; ++mt) acc[mt] = (v8f){};
#pragma unroll
    for (int ks = 0; ks < 4; ++ks) {
      const v16h bf = load_b_sw(ew1pL, ks, nt, NTE1);
#pragma unroll
      for (int mt = 0; mt < CHUNKM; ++mt) {
        const v16h af = load_a_edge(fi_sh, fj_sh, mt * 16, ks * 32);
        acc[mt] = WMMA_F16(af, bf, acc[mt]);
      }
    }
    const int n = nt * 16 + (lane & 15);
    const float w1d = ew1dpL[n];
    const float b1  = eb1pL[n];
#pragma unroll
    for (int mt = 0; mt < CHUNKM; ++mt) {
      const int mbase = mt * 16 + ((lane >> 4) << 3);
#pragma unroll
      for (int r = 0; r < 8; ++r) {
        const int m = mbase + r;
        h1_sh[m * H1P + n] = (half_t)silu_f(acc[mt][r] + dist_sh[m] * w1d + b1);
      }
    }
  }
  __syncthreads();

  // ---- GEMM2: h1 (Mp x 288pad) @ ew2 (258 x 32, padded) -> m_ij, silu, masked; + col sums ----
  for (int t = wv; t < CHUNKM * 2; t += 8) {
    const int mt = t >> 1, nt = t & 1;
    v8f acc = {};
#pragma unroll
    for (int ks = 0; ks < 9; ++ks) {
      const v16h af = load_a_lds(h1_sh, H1P, mt * 16, ks * 32);
      const v16h bf = load_b_sw(ew2pL, ks, nt, 2);
      acc = WMMA_F16(af, bf, acc);
    }
    const int n = nt * 16 + (lane & 15);
    const float b2 = eb2L[n];
    const int mbase = mt * 16 + ((lane >> 4) << 3);
    float colp = 0.0f;
#pragma unroll
    for (int r = 0; r < 8; ++r) {
      const int m = mbase + r;
      float v = silu_f(acc[r] + b2);
      if (!FULL && (j0 + m >= NN)) v = 0.0f;   // mask padded edges
      mij_sh[m * MDIMF + n] = (half_t)v;
      colp += v;
    }
    colp += __shfl_xor(colp, 16, 32);          // combine row halves (same n)
    if (lane < 16) atomicAdd(&msum[n], colp);
  }
  __syncthreads();

  // ---- GEMM3: m_ij (Mp x 32) @ cw1 (32 x 128), silu, dot cw2 -> w ----
#pragma unroll
  for (int mt = 0; mt < CHUNKM; ++mt) {
    const int nt = wv;                         // 8 waves cover all 8 n-tiles
    v8f acc = {};
    {
      const v16h af = load_a_lds(mij_sh, MDIMF, mt * 16, 0);
      const v16h bf = load_b_sw(cw1pL, 0, nt, 8);
      acc = WMMA_F16(af, bf, acc);
    }
    const int n = nt * 16 + (lane & 15);
    const float b = cb1L[n];
    const float c2 = cw2L[n];
    const int mbase = mt * 16 + ((lane >> 4) << 3);
    float rp[8];
#pragma unroll
    for (int r = 0; r < 8; ++r) rp[r] = silu_f(acc[r] + b) * c2;
#pragma unroll
    for (int off = 1; off < 16; off <<= 1) {
#pragma unroll
      for (int r = 0; r < 8; ++r) rp[r] += __shfl_xor(rp[r], off, 32);
    }
    if ((lane & 15) == 0) {
#pragma unroll
      for (int r = 0; r < 8; ++r) atomicAdd(&w_sh[mbase + r], rp[r]);
    }
  }
  __syncthreads();

  // ---- reductions out: m_i and coordinate delta ----
  if (tid < MDIMF)
    atomicAdd(&m_i[(size_t)(g * NN + node_i) * MDIMF + tid], msum[tid]);
  if (tid < Mp) {
    const int j = j0 + tid;
    if (FULL || j < NN) {
      const float d = dist_sh[tid];
      const float inv = 1.0f / fmaxf(sqrtf(d), 1e-8f);
      const float s = w_sh[tid] * (1.0f / 50.0f) * beta * inv;
      atomicAdd(&cdsum[0], s * rel_sh[tid * 3 + 0]);
      atomicAdd(&cdsum[1], s * rel_sh[tid * 3 + 1]);
      atomicAdd(&cdsum[2], s * rel_sh[tid * 3 + 2]);
    }
  }
  __syncthreads();
  if (tid < 3)
    atomicAdd(&cdelta[(size_t)(g * NN + node_i) * 3 + tid], cdsum[tid]);
}

// ---------------- node MLP (residual) + coordinate update ----------------
__global__ __launch_bounds__(256) void egnn_node_kernel(
    const float* __restrict__ feats, const float* __restrict__ coords,
    const float* __restrict__ m_i, const float* __restrict__ cdelta,
    const half_t* __restrict__ nw1p, const float* __restrict__ nb1,
    const half_t* __restrict__ nw2p, const float* __restrict__ nb2,
    float* __restrict__ feats_out, float* __restrict__ coords_out,
    int layer, int R) {
  __shared__ half_t x_sh[64 * 96];
  __shared__ half_t h_sh[64 * 128];
  const int tid = threadIdx.x;
  const int lane = tid & 31;
  const int wv = tid >> 5;
  const int row0 = blockIdx.x * 64;

  const half_t* nw1pL = nw1p + (size_t)layer * (3 * 8 * 512);
  const float*  nb1L = nb1 + (size_t)layer * 128;
  const half_t* nw2pL = nw2p + (size_t)layer * (4 * 4 * 512);
  const float*  nb2L = nb2 + (size_t)layer * 64;

  for (int idx = tid; idx < 64 * 96; idx += 256) {
    const int m = idx / 96, k = idx % 96;
    const int row = row0 + m;
    float v = 0.0f;
    if (row < R) v = (k < DIMF) ? feats[(size_t)row * DIMF + k]
                                : m_i[(size_t)row * MDIMF + (k - DIMF)];
    x_sh[m * 96 + k] = (half_t)v;
  }
  __syncthreads();

  // GEMM1: X (64x96) @ nw1 (96x128) -> h, silu ; wave w owns n-tile w, m-tiles 0..3
#pragma unroll
  for (int mt = 0; mt < 4; ++mt) {
    const int nt = wv;
    v8f acc = {};
#pragma unroll
    for (int ks = 0; ks < 3; ++ks) {
      const v16h af = load_a_lds(x_sh, 96, mt * 16, ks * 32);
      const v16h bf = load_b_sw(nw1pL, ks, nt, 8);
      acc = WMMA_F16(af, bf, acc);
    }
    const int n = nt * 16 + (lane & 15);
    const float b = nb1L[n];
    const int mbase = mt * 16 + ((lane >> 4) << 3);
#pragma unroll
    for (int r = 0; r < 8; ++r)
      h_sh[(mbase + r) * 128 + n] = (half_t)silu_f(acc[r] + b);
  }
  __syncthreads();

  // GEMM2: h (64x128) @ nw2 (128x64) + bias + residual ; 16 tiles over 8 waves
#pragma unroll
  for (int half = 0; half < 2; ++half) {
    const int t = wv + half * 8;
    const int nt = t & 3, mt = t >> 2;
    v8f acc = {};
#pragma unroll
    for (int ks = 0; ks < 4; ++ks) {
      const v16h af = load_a_lds(h_sh, 128, mt * 16, ks * 32);
      const v16h bf = load_b_sw(nw2pL, ks, nt, 4);
      acc = WMMA_F16(af, bf, acc);
    }
    const int n = nt * 16 + (lane & 15);
    const float b = nb2L[n];
    const int mbase = mt * 16 + ((lane >> 4) << 3);
#pragma unroll
    for (int r = 0; r < 8; ++r) {
      const int row = row0 + mbase + r;
      if (row < R)
        feats_out[(size_t)row * DIMF + n] = acc[r] + b + feats[(size_t)row * DIMF + n];
    }
  }

  // coords update
  for (int idx = tid; idx < 64 * 3; idx += 256) {
    const int m = idx / 3, c = idx % 3;
    const int row = row0 + m;
    if (row < R)
      coords_out[(size_t)row * 3 + c] = coords[(size_t)row * 3 + c] + cdelta[(size_t)row * 3 + c];
  }
}

// ---------------- layout transpose A:(b,l,n,C) <-> B:(b,n,l,C) ----------------
__global__ void transpose_kernel(const float* __restrict__ in, float* __restrict__ out,
                                 int Bb, int L, int Nn, int C, int a2b) {
  const int total = Bb * L * Nn * C;
  for (int idx = blockIdx.x * blockDim.x + threadIdx.x; idx < total;
       idx += gridDim.x * blockDim.x) {
    const int c = idx % C;
    int rest = idx / C;
    const int n = rest % Nn; rest /= Nn;
    const int l = rest % L;
    const int b = rest / L;
    const int idxB = (((b * Nn + n) * L + l) * C + c);
    if (a2b) out[idxB] = in[idx];
    else     out[idx]  = in[idxB];
  }
}

// ---------------- host orchestration ----------------
extern "C" void kernel_launch(void* const* d_in, const int* in_sizes, int n_in,
                              void* d_out, int out_size, void* d_ws, size_t ws_size,
                              hipStream_t stream) {
  const int*   tokens  = (const int*)d_in[0];
  const float* cords   = (const float*)d_in[1];
  const float* tok_emb = (const float*)d_in[2];
  const float* pos_emb = (const float*)d_in[3];
  const float* ew1 = (const float*)d_in[4];
  const float* eb1 = (const float*)d_in[5];
  const float* ew2 = (const float*)d_in[6];
  const float* eb2 = (const float*)d_in[7];
  const float* cw1 = (const float*)d_in[8];
  const float* cb1 = (const float*)d_in[9];
  const float* cw2 = (const float*)d_in[10];
  const float* cb2 = (const float*)d_in[11];
  const float* nw1 = (const float*)d_in[12];
  const float* nb1 = (const float*)d_in[13];
  const float* nw2 = (const float*)d_in[14];
  const float* nb2 = (const float*)d_in[15];
  const float* ln_b = (const float*)d_in[17];   // ln_g (d_in[16]) is dead: g*0+beta
  (void)in_sizes; (void)n_in; (void)out_size; (void)ws_size;

  const int R = RTOT;
  float* ws = (float*)d_ws;
  float* feats0  = ws;                                   // R*64
  float* feats1  = feats0 + (size_t)R * DIMF;            // R*64
  float* coords0 = feats1 + (size_t)R * DIMF;            // R*3
  float* coords1 = coords0 + (size_t)R * 3;              // R*3
  float* m_i     = coords1 + (size_t)R * 3;              // R*32
  float* cdelta  = m_i + (size_t)R * MDIMF;              // R*3
  float* eb1p    = cdelta + (size_t)R * 3;               // 16*272
  float* ew1dp   = eb1p + NLAYER * (NTE1 * 16);          // 16*272
  half_t* packs  = (half_t*)(ew1dp + NLAYER * (NTE1 * 16));
  half_t* ew1p = packs;                                  // 16 * 4*17*512
  half_t* ew2p = ew1p + (size_t)NLAYER * (4 * NTE1 * 512);   // 16 * 9*2*512
  half_t* cw1p = ew2p + (size_t)NLAYER * (9 * 2 * 512);      // 16 * 1*8*512
  half_t* nw1p = cw1p + (size_t)NLAYER * (1 * 8 * 512);      // 16 * 3*8*512
  half_t* nw2p = nw1p + (size_t)NLAYER * (3 * 8 * 512);      // 16 * 4*4*512

  // ---- one-time (per launch) weight packing ----
  {
    dim3 g16(32, NLAYER);
    pack_b_kernel<<<g16, 256, 0, stream>>>(ew1, ew1p, EINF, H1F, 128, 4, NTE1);
    pack_b_kernel<<<g16, 256, 0, stream>>>(ew2, ew2p, H1F, MDIMF, H1F, 9, 2);
    pack_b_kernel<<<g16, 256, 0, stream>>>(cw1, cw1p, MDIMF, 128, MDIMF, 1, 8);
    pack_b_kernel<<<g16, 256, 0, stream>>>(nw1, nw1p, 96, 128, 96, 3, 8);
    pack_b_kernel<<<g16, 256, 0, stream>>>(nw2, nw2p, 128, DIMF, 128, 4, 4);
    dim3 g1(2, NLAYER);
    pad_vec_kernel<<<g1, 256, 0, stream>>>(eb1, eb1p, H1F, NTE1 * 16, H1F);
    pad_vec_kernel<<<g1, 256, 0, stream>>>(ew1 + 128 * H1F, ew1dp, H1F, NTE1 * 16, EINF * H1F);
  }

  embed_kernel<<<192, 256, 0, stream>>>(tokens, cords, tok_emb, pos_emb, feats0, coords0);

  int cur = 0;
  for (int layer = 0; layer < 16; ++layer) {
    const int group = layer >> 2;
    if (layer > 0 && (layer & 3) == 0) {
      const int a2b = (group & 1);  // group1: A->B, group2: B->A, group3: A->B
      float* fin  = cur ? feats1 : feats0;
      float* fout = cur ? feats0 : feats1;
      float* cin  = cur ? coords1 : coords0;
      float* cout = cur ? coords0 : coords1;
      transpose_kernel<<<(R * DIMF + 255) / 256, 256, 0, stream>>>(fin, fout, 2, 96, 23, DIMF, a2b);
      transpose_kernel<<<(R * 3 + 255) / 256, 256, 0, stream>>>(cin, cout, 2, 96, 23, 3, a2b);
      cur ^= 1;
    }
    const bool layoutA = ((group & 1) == 0);

    float* fcur = cur ? feats1 : feats0;
    float* fnxt = cur ? feats0 : feats1;
    float* ccur = cur ? coords1 : coords0;
    float* cnxt = cur ? coords0 : coords1;

    zero_kernel<<<(R * MDIMF + 255) / 256, 256, 0, stream>>>(m_i, R * MDIMF);
    zero_kernel<<<(R * 3 + 255) / 256, 256, 0, stream>>>(cdelta, R * 3);

    if (layoutA) {
      egnn_edge_kernel<23, 1, 2><<<192 * 23 * 1, 256, 0, stream>>>(
          fcur, ccur, ew1p, eb1p, ew1dp, ew2p, eb2, cw1p, cb1, cw2, cb2, ln_b,
          m_i, cdelta, layer);
    } else {
      egnn_edge_kernel<96, 2, 3><<<46 * 96 * 2, 256, 0, stream>>>(
          fcur, ccur, ew1p, eb1p, ew1dp, ew2p, eb2, cw1p, cb1, cw2, cb2, ln_b,
          m_i, cdelta, layer);
    }

    egnn_node_kernel<<<(R + 63) / 64, 256, 0, stream>>>(
        fcur, ccur, m_i, cdelta, nw1p, nb1, nw2p, nb2, fnxt, cnxt, layer, R);

    cur ^= 1;
  }

  // final: coords are in layout B; reference swaps back to (b, l*N, 3)
  float* cfin = cur ? coords1 : coords0;
  transpose_kernel<<<(R * 3 + 255) / 256, 256, 0, stream>>>(
      cfin, (float*)d_out, 2, 96, 23, 3, /*a2b=*/0);
}
